// NodeModel_49546742726708
// MI455X (gfx1250) — matrix operand order repaired and checked by reading, not compiled, not run on hardware
//
#include <hip/hip_runtime.h>

// MI455X / gfx1250: wave32, WMMA 16x16x32 f16 -> f32.
// Edge MLP + scatter-add + Node MLP, all GEMM work on v_wmma_f32_16x16x32_f16.

typedef __attribute__((ext_vector_type(16))) _Float16 v16h;
typedef __attribute__((ext_vector_type(8)))  float    v8f;

#define H 128          // feature dim (also padded hidden dim)

// ---- CDNA5 ISA 7.12.2 fragment index maps (16-bit data, wave32) ----
// A (16x32, MxK): lane = (m&15) + 16*((kk>>3)&1); vgpr j = 4*(kk>>4) + ((kk&7)>>1); half = kk&1
__device__ inline int a_half_off(int m, int k) {
  int kk = k & 31;
  int lane = (m & 15) + 16 * ((kk >> 3) & 1);
  int j = 4 * (kk >> 4) + ((kk & 7) >> 1);
  return (k >> 5) * 512 + lane * 16 + j * 2 + (kk & 1);
}
// B (32x16, KxN): lane = (n&15) + 16*((kk>>3)&1); vgpr j = 4*(kk>>4) + ((kk&7)>>1); half = kk&1
__device__ inline int b_half_off(int k, int n, int numKt) {
  int kk = k & 31;
  int lane = (n & 15) + 16 * ((kk >> 3) & 1);
  int j = 4 * (kk >> 4) + ((kk & 7) >> 1);
  int frag = (n >> 4) * numKt + (k >> 5);
  return frag * 512 + lane * 16 + j * 2 + (kk & 1);
}

// ---- one-time weight repack: fp32 [finReal, foutReal] -> f16 B-fragments, zero padded ----
__global__ void pack_w(const float* __restrict__ w, int finReal, int foutReal,
                       int finPad, _Float16* __restrict__ dst) {
  int total = finPad * H;
  int numKt = finPad >> 5;
  for (int idx = blockIdx.x * blockDim.x + threadIdx.x; idx < total;
       idx += gridDim.x * blockDim.x) {
    int k = idx >> 7;      // 0..finPad-1
    int n = idx & (H - 1); // 0..127
    float v = (k < finReal && n < foutReal) ? w[k * foutReal + n] : 0.f;
    dst[b_half_off(k, n, numKt)] = (_Float16)v;
  }
}

__global__ void pack_bias(const float* __restrict__ b, int nReal, float* __restrict__ dst) {
  int n = threadIdx.x;
  if (n < H) dst[n] = (n < nReal) ? b[n] : 0.f;
}

__global__ void zero_f(float* __restrict__ p, int n) {
  for (int i = blockIdx.x * blockDim.x + threadIdx.x; i < n; i += gridDim.x * blockDim.x)
    p[i] = 0.f;
}

// ---- one MLP layer on a 16-row tile, all-WMMA ----
// inBuf: A-fragment-packed f16 activations in wave-private LDS (NKT k-tiles).
// W: packed B fragments (global, L2-resident). bias: 128 fp32.
// RELU: relu + store f16 in A-layout to outBuf; else raw f32 to outF[16][128].
template <int NKT, bool RELU>
__device__ inline void mlp_layer(const _Float16* __restrict__ inBuf,
                                 const _Float16* __restrict__ W,
                                 const float* __restrict__ bias,
                                 _Float16* __restrict__ outBuf,
                                 float* __restrict__ outF, int lane) {
  v16h afr[NKT];
#pragma unroll
  for (int kt = 0; kt < NKT; ++kt)
    afr[kt] = *(const v16h*)(inBuf + kt * 512 + lane * 16);

  const int mrow = 8 * (lane >> 4);
#pragma unroll
  for (int nt = 0; nt < 8; ++nt) {
    float bv = bias[nt * 16 + (lane & 15)];
    v8f acc = {bv, bv, bv, bv, bv, bv, bv, bv};
#pragma unroll
    for (int kt = 0; kt < NKT; ++kt) {
      v16h bfr = *(const v16h*)(W + (nt * NKT + kt) * 512 + lane * 16);
      acc = __builtin_amdgcn_wmma_f32_16x16x32_f16(
          false, afr[kt], false, bfr, (short)0, acc, false, false);
    }
    int n = nt * 16 + (lane & 15);
#pragma unroll
    for (int r = 0; r < 8; ++r) {
      float v = acc[r];
      if (RELU) {
        v = v > 0.f ? v : 0.f;
        outBuf[a_half_off(r + mrow, n)] = (_Float16)v;  // C-layout -> A-layout transpose via LDS
      } else {
        outF[(r + mrow) * H + n] = v;
      }
    }
  }
  // wave-private LDS: only need DS-counter ordering, no barrier
  asm volatile("s_wait_dscnt 0" ::: "memory");
}

// ---- edge MLP + LayerNorm + atomic scatter-add into agg (= d_out) ----
__global__ __launch_bounds__(128) void edge_kernel(
    const float* __restrict__ x, const int* __restrict__ ei,
    const float* __restrict__ ea, const _Float16* __restrict__ wp,
    const float* __restrict__ bp, const float* __restrict__ gma,
    const float* __restrict__ bta, float* __restrict__ agg, int E) {
  __shared__ _Float16 sA[4][16 * 256];  // 8KB/wave
  __shared__ _Float16 sB[4][16 * 128];  // 4KB/wave
  const int lane = threadIdx.x & 31;
  const int wave = threadIdx.x >> 5;
  _Float16* bufA = sA[wave];
  _Float16* bufB = sB[wave];
  float* bufF = (float*)sA[wave];
  const int edge0 = (blockIdx.x * 4 + wave) * 16;

  // stage [16 x 256] = [x[row] | edge_attr], coalesced, f32->f16, A-layout
#pragma unroll 4
  for (int i = 0; i < 128; ++i) {
    int flat = i * 32 + lane;
    int m = flat >> 8;
    int k = flat & 255;
    int e = edge0 + m;
    int ec = e < E ? e : 0;
    float v;
    if (k < H) v = x[(long)ei[ec] * H + k];
    else       v = ea[(long)ec * H + (k - H)];
    bufA[a_half_off(m, k)] = (_Float16)(e < E ? v : 0.f);
  }
  asm volatile("s_wait_dscnt 0" ::: "memory");

  mlp_layer<8, true >(bufA, wp + 0,     bp + 0 * H, bufB, nullptr, lane);
  mlp_layer<4, true >(bufB, wp + 32768, bp + 1 * H, bufA, nullptr, lane);
  mlp_layer<4, true >(bufA, wp + 49152, bp + 2 * H, bufB, nullptr, lane);
  mlp_layer<4, false>(bufB, wp + 65536, bp + 3 * H, nullptr, bufF, lane);

  // LayerNorm per edge row (2 lanes/row), then atomic scatter to agg[col]
  const int m = lane & 15;
  const int hh = lane >> 4;
  float s = 0.f, s2 = 0.f;
#pragma unroll 8
  for (int i = 0; i < 64; ++i) {
    float v = bufF[m * H + hh * 64 + i];
    s += v; s2 += v * v;
  }
  s  += __shfl_xor(s, 16, 32);
  s2 += __shfl_xor(s2, 16, 32);
  float mean = s * (1.f / 128.f);
  float var  = s2 * (1.f / 128.f) - mean * mean;
  float rstd = rsqrtf(var + 1e-5f);
  int e = edge0 + m;
  if (e < E) {
    long dst = (long)ei[E + e] * H;
#pragma unroll 4
    for (int i = 0; i < 64; ++i) {
      int n = hh * 64 + i;
      float v = (bufF[m * H + n] - mean) * rstd * gma[n] + bta[n];
      atomicAdd(&agg[dst + n], v);  // global_atomic_add_f32, L2-resident target
    }
  }
}

// ---- node MLP: [x | agg | u[batch]] -> LN -> residual; agg lives in `out` ----
__global__ __launch_bounds__(128) void node_kernel(
    const float* __restrict__ x, const float* __restrict__ u,
    const int* __restrict__ batch, const _Float16* __restrict__ wp,
    const float* __restrict__ bp, const float* __restrict__ gma,
    const float* __restrict__ bta, float* __restrict__ out, int N) {
  __shared__ _Float16 sA[4][16 * 384];  // 12KB/wave
  __shared__ _Float16 sB[4][16 * 128];  // 4KB/wave  (total 64KB)
  const int lane = threadIdx.x & 31;
  const int wave = threadIdx.x >> 5;
  _Float16* bufA = sA[wave];
  _Float16* bufB = sB[wave];
  float* bufF = (float*)sA[wave];
  const int row0 = (blockIdx.x * 4 + wave) * 16;

  // stage [16 x 384] = [x | agg(=out) | u[batch]]; each wave reads only its own rows of `out`
#pragma unroll 4
  for (int i = 0; i < 192; ++i) {
    int flat = i * 32 + lane;
    int m = flat / 384;
    int k = flat % 384;
    int r = row0 + m;
    int rc = r < N ? r : 0;
    float v;
    if      (k < H)     v = x[(long)rc * H + k];
    else if (k < 2 * H) v = out[(long)rc * H + (k - H)];
    else                v = u[(long)batch[rc] * H + (k - 2 * H)];
    bufA[a_half_off(m, k)] = (_Float16)(r < N ? v : 0.f);
  }
  asm volatile("s_wait_dscnt 0" ::: "memory");

  mlp_layer<12, true>(bufA, wp + 0,     bp + 0 * H, bufB, nullptr, lane);
  mlp_layer<4,  true>(bufB, wp + 49152, bp + 1 * H, bufA, nullptr, lane);
  mlp_layer<4,  true>(bufA, wp + 65536, bp + 2 * H, bufB, nullptr, lane);
  mlp_layer<4, false>(bufB, wp + 81920, bp + 3 * H, nullptr, bufF, lane);

  const int m = lane & 15;
  const int hh = lane >> 4;
  float s = 0.f, s2 = 0.f;
#pragma unroll 8
  for (int i = 0; i < 64; ++i) {
    float v = bufF[m * H + hh * 64 + i];
    s += v; s2 += v * v;
  }
  s  += __shfl_xor(s, 16, 32);
  s2 += __shfl_xor(s2, 16, 32);
  float mean = s * (1.f / 128.f);
  float var  = s2 * (1.f / 128.f) - mean * mean;
  float rstd = rsqrtf(var + 1e-5f);
  int r = row0 + m;
  if (r < N) {
#pragma unroll 4
    for (int i = 0; i < 64; ++i) {
      int n = hh * 64 + i;
      float v = (bufF[m * H + n] - mean) * rstd * gma[n] + bta[n];
      out[(long)r * H + n] = x[(long)r * H + n] + v;  // residual
    }
  }
}

extern "C" void kernel_launch(void* const* d_in, const int* in_sizes, int n_in,
                              void* d_out, int out_size, void* d_ws, size_t ws_size,
                              hipStream_t stream) {
  const float* x    = (const float*)d_in[0];
  const int*   ei   = (const int*)d_in[1];
  const float* ea   = (const float*)d_in[2];
  const float* u    = (const float*)d_in[3];
  const int*   batch= (const int*)d_in[4];
  const float* m1_w[4] = {(const float*)d_in[5],  (const float*)d_in[7],
                          (const float*)d_in[9],  (const float*)d_in[11]};
  const float* m1_b[4] = {(const float*)d_in[6],  (const float*)d_in[8],
                          (const float*)d_in[10], (const float*)d_in[12]};
  const float* m1_g  = (const float*)d_in[13];
  const float* m1_be = (const float*)d_in[14];
  const float* m2_w[4] = {(const float*)d_in[15], (const float*)d_in[17],
                          (const float*)d_in[19], (const float*)d_in[21]};
  const float* m2_b[4] = {(const float*)d_in[16], (const float*)d_in[18],
                          (const float*)d_in[20], (const float*)d_in[22]};
  const float* m2_g  = (const float*)d_in[23];
  const float* m2_be = (const float*)d_in[24];

  const int N = in_sizes[0] / H;     // 50000
  const int E = in_sizes[1] / 2;     // 800000
  float* out = (float*)d_out;

  // workspace layout (halves): m1 L0..L3 at 0/32768/49152/65536; m2 L0..L3 at 81920 + 0/49152/65536/81920
  _Float16* wp = (_Float16*)d_ws;
  float* bp = (float*)((char*)d_ws + 180224 * sizeof(_Float16));

  // --- pack weights (tiny, L2 resident afterwards) ---
  pack_w<<<128, 256, 0, stream>>>(m1_w[0], 256, 100, 256, wp + 0);
  pack_w<<<64,  256, 0, stream>>>(m1_w[1], 100, 100, 128, wp + 32768);
  pack_w<<<64,  256, 0, stream>>>(m1_w[2], 100, 100, 128, wp + 49152);
  pack_w<<<64,  256, 0, stream>>>(m1_w[3], 100, 128, 128, wp + 65536);
  pack_w<<<192, 256, 0, stream>>>(m2_w[0], 384, 100, 384, wp + 81920);
  pack_w<<<64,  256, 0, stream>>>(m2_w[1], 100, 100, 128, wp + 81920 + 49152);
  pack_w<<<64,  256, 0, stream>>>(m2_w[2], 100, 100, 128, wp + 81920 + 65536);
  pack_w<<<64,  256, 0, stream>>>(m2_w[3], 100, 128, 128, wp + 81920 + 81920);
  for (int l = 0; l < 4; ++l) {
    pack_bias<<<1, 128, 0, stream>>>(m1_b[l], (l == 3) ? 128 : 100, bp + l * H);
    pack_bias<<<1, 128, 0, stream>>>(m2_b[l], (l == 3) ? 128 : 100, bp + (4 + l) * H);
  }

  // --- zero d_out (doubles as the agg accumulator) ---
  zero_f<<<1024, 256, 0, stream>>>(out, N * H);

  // --- edge MLP + scatter ---
  edge_kernel<<<(E + 63) / 64, 128, 0, stream>>>(x, ei, ea, wp, bp, m1_g, m1_be, out, E);

  // --- node MLP + residual (reads agg from its own rows of out, then overwrites) ---
  node_kernel<<<(N + 63) / 64, 128, 0, stream>>>(x, u, batch, wp + 81920, bp + 4 * H,
                                                 m2_g, m2_be, out, N);
}
